// GCNEncoder_26113401160170
// MI455X (gfx1250) — matrix operand order repaired and checked by reading, not compiled, not run on hardware
//
#include <hip/hip_runtime.h>
#include <hip/hip_bf16.h>

#define NB   256   // graphs
#define NM   512   // nodes per graph
#define NK   6     // knn k
#define NC   32    // channels

typedef __attribute__((ext_vector_type(16))) _Float16 v16h;
typedef __attribute__((ext_vector_type(8)))  _Float16 v8h;
typedef __attribute__((ext_vector_type(8)))  float    v8f;

// ---------------------------------------------------------------------------
// WMMA fragment helpers (gfx1250 wave32 layouts, CDNA5 ISA 7.12.2)
// All 16-f16 halves are contiguous 16B chunks -> ds_load_b128/global_load_b128.
// ---------------------------------------------------------------------------

__device__ inline v16h merge16(v8h lo, v8h hi) {
    return __builtin_shufflevector(lo, hi, 0, 1, 2, 3, 4, 5, 6, 7,
                                           8, 9, 10, 11, 12, 13, 14, 15);
}

// A matrix 16x32 f16 from an LDS tile stored row-major [16][32] (64B rows).
// lanes 0-15: row=lane,    K 0..15 ; lanes 16-31: row=lane-16, K 16..31
__device__ inline v16h load_a_frag(const _Float16* tile, int lane) {
    int row = lane & 15;
    int kb  = (lane & 16) ? 16 : 0;
    const _Float16* p = tile + row * NC;
    v8h lo = *(const v8h*)(p + kb);
    v8h hi = *(const v8h*)(p + 8 + kb);
    return merge16(lo, hi);
}

// B matrix 32x16 f16 = W^T half (out-channel block half*16..+15), W f16 row-major
// [32 out][32 in] in GLOBAL memory (pre-converted, fully L2-resident).
// B[k][n] = W[n + 16*half][k]; lanes 0-15 K=0..15, lanes 16-31 K=16..31.
__device__ inline v16h load_b_frag(const _Float16* __restrict__ W, int lane, int half) {
    int n  = (lane & 15) + 16 * half;
    int kb = (lane & 16) ? 16 : 0;
    const v8h* p = (const v8h*)(W + n * NC + kb);
    return merge16(p[0], p[1]);
}

// C/D 16x16 f32: lanes 0-15 VGPR r -> (M=r, N=lane); lanes 16-31 -> (M=8+r, N=lane-16).
// N constant per lane => bias splat-seeded into the accumulator.
__device__ inline v8f bias_acc(const float* __restrict__ bias, int lane, int half) {
    float bv = bias[(lane & 15) + 16 * half];
    v8f c;
#pragma unroll
    for (int r = 0; r < 8; ++r) c[r] = bv;
    return c;
}

__device__ inline void store_d_f16_relu(_Float16* tile, v8f acc, int lane, int half) {
    int n  = (lane & 15) + 16 * half;
    int mb = (lane & 16) ? 8 : 0;
#pragma unroll
    for (int r = 0; r < 8; ++r)
        tile[(mb + r) * NC + n] = (_Float16)fmaxf(acc[r], 0.0f);
}

__device__ inline void store_d_f32(float* buf, v8f acc, int lane, int half) {
    int n  = (lane & 15) + 16 * half;
    int mb = (lane & 16) ? 8 : 0;
#pragma unroll
    for (int r = 0; r < 8; ++r)
        buf[(mb + r) * NC + n] = acc[r];
}

// ---------------------------------------------------------------------------
// Kernel 1: per-graph KNN (top-6 including self), positions staged in LDS
// ---------------------------------------------------------------------------
__global__ __launch_bounds__(256)
void knn_kernel(const float* __restrict__ pos, int* __restrict__ idx) {
    int b = blockIdx.x;
    __shared__ float sx[NM], sy[NM], sz[NM];
    for (int i = threadIdx.x; i < NM; i += blockDim.x) {
        const float* p = pos + ((size_t)b * NM + i) * 3;
        sx[i] = p[0]; sy[i] = p[1]; sz[i] = p[2];
    }
    __syncthreads();
    for (int m = threadIdx.x; m < NM; m += blockDim.x) {
        float px = sx[m], py = sy[m], pz = sz[m];
        float bd[NK]; int bi[NK];
#pragma unroll
        for (int k = 0; k < NK; ++k) { bd[k] = 3.4e38f; bi[k] = m; }
        for (int j = 0; j < NM; ++j) {
            float dx = sx[j] - px, dy = sy[j] - py, dz = sz[j] - pz;
            float d = dx * dx + dy * dy + dz * dz;
            if (d < bd[NK - 1]) {
                int k = NK - 1;
                while (k > 0 && bd[k - 1] > d) { bd[k] = bd[k - 1]; bi[k] = bi[k - 1]; --k; }
                bd[k] = d; bi[k] = j;
            }
        }
#pragma unroll
        for (int k = 0; k < NK; ++k)
            idx[((size_t)b * NM + m) * NK + k] = bi[k];
    }
}

// ---------------------------------------------------------------------------
// Kernel 2: h0 = concat(x, pos) zero-padded to 32 channels (f16), and
// one-time f32->f16 weight prep (W1a zero-padded 4->32 input channels).
// ---------------------------------------------------------------------------
__global__ __launch_bounds__(256)
void build_h0(const float* __restrict__ x, const float* __restrict__ pos,
              _Float16* __restrict__ h0, int n_nodes) {
    int n = blockIdx.x * blockDim.x + threadIdx.x;
    if (n >= n_nodes) return;
    _Float16* r = h0 + (size_t)n * NC;
    r[0] = (_Float16)x[n];
    r[1] = (_Float16)pos[n * 3 + 0];
    r[2] = (_Float16)pos[n * 3 + 1];
    r[3] = (_Float16)pos[n * 3 + 2];
#pragma unroll
    for (int c = 4; c < NC; ++c) r[c] = (_Float16)0.0f;
}

__global__ __launch_bounds__(256)
void prep_weights(const float* __restrict__ W1a, const float* __restrict__ W2a,
                  const float* __restrict__ W1b, const float* __restrict__ W2b,
                  const float* __restrict__ W1c, const float* __restrict__ W2c,
                  _Float16* __restrict__ wf) {
    int i = blockIdx.x * blockDim.x + threadIdx.x;
    if (i >= NC * NC) return;
    int o = i >> 5, c = i & 31;
    wf[0 * NC * NC + i] = (c < 4) ? (_Float16)W1a[o * 4 + c] : (_Float16)0.0f;
    wf[1 * NC * NC + i] = (_Float16)W2a[i];
    wf[2 * NC * NC + i] = (_Float16)W1b[i];
    wf[3 * NC * NC + i] = (_Float16)W2b[i];
    wf[4 * NC * NC + i] = (_Float16)W1c[i];
    wf[5 * NC * NC + i] = (_Float16)W2c[i];
}

// ---------------------------------------------------------------------------
// Kernel 3: one GCN layer.  Block = 32 nodes = 192 edges = 12 WMMA tiles,
// 128 threads = 4 waves, 3 tiles per wave.  Weights/biases read from global
// (L2-resident); activations staged in LDS.
// ---------------------------------------------------------------------------
#define NODES_BLK 32
#define EDGES_BLK (NODES_BLK * NK)   // 192
#define TILES_BLK (EDGES_BLK / 16)   // 12

__global__ __launch_bounds__(128)
void gcn_layer_kernel(const _Float16* __restrict__ hsrc, _Float16* __restrict__ hdst,
                      const int* __restrict__ idx,
                      const _Float16* __restrict__ W1f, const float* __restrict__ b1,
                      const _Float16* __restrict__ W2f, const float* __restrict__ b2) {
    __shared__ __align__(16) _Float16 sFeat[EDGES_BLK * NC];  // 12 KB gathered edges
    __shared__ __align__(16) _Float16 sMsg[4][16 * NC];       // 4 KB per-wave staging
    __shared__ float sOut[EDGES_BLK * NC];                    // 24 KB msg2 f32 for K-max

    int tid      = threadIdx.x;
    int b        = blockIdx.x >> 4;                 // NM / NODES_BLK = 16 blocks/graph
    int nodeBase = (blockIdx.x & 15) * NODES_BLK;

    // Gather neighbor feature rows: 192 edges x 64 bytes, uint4 chunks.
    for (int c = tid; c < EDGES_BLK * 4; c += blockDim.x) {
        int e = c >> 2, part = c & 3;
        int node = nodeBase + e / NK;
        int k    = e - (e / NK) * NK;
        int j    = idx[((size_t)b * NM + node) * NK + k];
        const uint4* src = (const uint4*)(hsrc + ((size_t)b * NM + j) * NC);
        ((uint4*)(sFeat + e * NC))[part] = src[part];
    }
    __syncthreads();

    // WMMA edge MLP: uniform per-wave control flow, EXEC all ones.
    int wave = tid >> 5;
    int lane = tid & 31;
    _Float16* mbuf = &sMsg[wave][0];

    v16h b1lo = load_b_frag(W1f, lane, 0);
    v16h b1hi = load_b_frag(W1f, lane, 1);
    v16h b2lo = load_b_frag(W2f, lane, 0);
    v16h b2hi = load_b_frag(W2f, lane, 1);
    v8f c1lo = bias_acc(b1, lane, 0);
    v8f c1hi = bias_acc(b1, lane, 1);
    v8f c2lo = bias_acc(b2, lane, 0);
    v8f c2hi = bias_acc(b2, lane, 1);

    for (int t = wave; t < TILES_BLK; t += 4) {
        v16h a = load_a_frag(sFeat + t * 16 * NC, lane);

        v8f acc0 = __builtin_amdgcn_wmma_f32_16x16x32_f16(false, a, false, b1lo,
                                                          (short)0, c1lo, false, false);
        v8f acc1 = __builtin_amdgcn_wmma_f32_16x16x32_f16(false, a, false, b1hi,
                                                          (short)0, c1hi, false, false);
        store_d_f16_relu(mbuf, acc0, lane, 0);
        store_d_f16_relu(mbuf, acc1, lane, 1);
        asm volatile("s_wait_dscnt 0" ::: "memory");   // intra-wave LDS RAW fence

        v16h a2 = load_a_frag(mbuf, lane);
        v8f o0 = __builtin_amdgcn_wmma_f32_16x16x32_f16(false, a2, false, b2lo,
                                                        (short)0, c2lo, false, false);
        v8f o1 = __builtin_amdgcn_wmma_f32_16x16x32_f16(false, a2, false, b2hi,
                                                        (short)0, c2hi, false, false);
        store_d_f32(sOut + t * 16 * NC, o0, lane, 0);
        store_d_f32(sOut + t * 16 * NC, o1, lane, 1);
    }
    __syncthreads();

    // Max over K=6 neighbors + outer relu, store f16.
    for (int i = tid; i < NODES_BLK * NC; i += blockDim.x) {
        int nloc = i >> 5, ch = i & 31;
        const float* p = sOut + (nloc * NK) * NC + ch;
        float m = p[0];
#pragma unroll
        for (int k = 1; k < NK; ++k) m = fmaxf(m, p[k * NC]);
        m = fmaxf(m, 0.0f);
        hdst[((size_t)b * NM + nodeBase + nloc) * NC + ch] = (_Float16)m;
    }
}

// ---------------------------------------------------------------------------
// Kernel 4: global max pool over nodes + 6x32 linear head (f32 out)
// ---------------------------------------------------------------------------
__global__ __launch_bounds__(128)
void pool_head(const _Float16* __restrict__ h, const float* __restrict__ Wr,
               const float* __restrict__ br, float* __restrict__ out) {
    int b = blockIdx.x;
    __shared__ float part[4][NC];
    __shared__ float g[NC];
    int t = threadIdx.x;
    int ch = t & 31, seg = t >> 5;
    float m = -3.4e38f;
    for (int n = seg * 128; n < seg * 128 + 128; ++n)
        m = fmaxf(m, (float)h[((size_t)b * NM + n) * NC + ch]);
    part[seg][ch] = m;
    __syncthreads();
    if (t < NC)
        g[t] = fmaxf(fmaxf(part[0][t], part[1][t]), fmaxf(part[2][t], part[3][t]));
    __syncthreads();
    if (t < 6) {
        float acc = br[t];
#pragma unroll
        for (int c = 0; c < NC; ++c) acc += g[c] * Wr[t * NC + c];
        out[b * 6 + t] = acc;
    }
}

// ---------------------------------------------------------------------------
extern "C" void kernel_launch(void* const* d_in, const int* in_sizes, int n_in,
                              void* d_out, int out_size, void* d_ws, size_t ws_size,
                              hipStream_t stream) {
    const float* x   = (const float*)d_in[0];
    const float* pos = (const float*)d_in[1];
    // d_in[2] = batch (unused; implied by contiguous [B, M] layout)
    const float* W1a = (const float*)d_in[3];  const float* b1a = (const float*)d_in[4];
    const float* W2a = (const float*)d_in[5];  const float* b2a = (const float*)d_in[6];
    const float* W1b = (const float*)d_in[7];  const float* b1b = (const float*)d_in[8];
    const float* W2b = (const float*)d_in[9];  const float* b2b = (const float*)d_in[10];
    const float* W1c = (const float*)d_in[11]; const float* b1c = (const float*)d_in[12];
    const float* W2c = (const float*)d_in[13]; const float* b2c = (const float*)d_in[14];
    const float* Wr  = (const float*)d_in[15]; const float* br  = (const float*)d_in[16];
    float* out = (float*)d_out;

    char* ws = (char*)d_ws;
    int*      idx = (int*)ws;                                  // 3 MB
    _Float16* hA  = (_Float16*)(ws + ((size_t)4  << 20));      // 8 MB
    _Float16* hB  = (_Float16*)(ws + ((size_t)13 << 20));      // 8 MB
    _Float16* wf  = (_Float16*)(ws + ((size_t)22 << 20));      // 12 KB f16 weights

    const int n_nodes = NB * NM;

    knn_kernel<<<NB, 256, 0, stream>>>(pos, idx);
    build_h0<<<(n_nodes + 255) / 256, 256, 0, stream>>>(x, pos, hA, n_nodes);
    prep_weights<<<(NC * NC + 255) / 256, 256, 0, stream>>>(W1a, W2a, W1b, W2b,
                                                            W1c, W2c, wf);

    dim3 ggrid(NB * (NM / NODES_BLK));
    const int WSZ = NC * NC;
    gcn_layer_kernel<<<ggrid, 128, 0, stream>>>(hA, hB, idx, wf + 0 * WSZ, b1a,
                                                wf + 1 * WSZ, b2a);
    gcn_layer_kernel<<<ggrid, 128, 0, stream>>>(hB, hA, idx, wf + 2 * WSZ, b1b,
                                                wf + 3 * WSZ, b2b);
    gcn_layer_kernel<<<ggrid, 128, 0, stream>>>(hA, hB, idx, wf + 4 * WSZ, b1c,
                                                wf + 5 * WSZ, b2c);

    pool_head<<<NB, 128, 0, stream>>>(hB, Wr, br, out);
}
